// NystromTransformerLayer_36919538876522
// MI455X (gfx1250) — matrix operand order repaired and checked by reading, not compiled, not run on hardware
//
#include <hip/hip_runtime.h>
#include <math.h>

// ---------------------------------------------------------------------------
// Nystromformer layer for MI455X (gfx1250, wave32, WMMA).
//   B=8 N=4096 D=512 H=8 DH=64 M=256 landmarks, 6 Newton-Schulz pinv iters,
//   depthwise conv K=33, MLP 512->2048->512 with exact GELU.
// All big matmuls on v_wmma_f32_16x16x32_f16 (activations) or
// v_wmma_f32_16x16x4_f32 (pinv, kept f32 for iteration stability).
// GEMMs are double-buffered in LDS (global loads for tile k+1 issued before
// the WMMA work on tile k) and templated on layout/epilogue so the k-loop is
// branch-free.  Workspace requirement: ~790 MB (lifetime-aliased regions).
// ---------------------------------------------------------------------------

typedef _Float16 v16h __attribute__((ext_vector_type(16)));
typedef _Float16 v8h  __attribute__((ext_vector_type(8)));
typedef float    v8f  __attribute__((ext_vector_type(8)));
typedef float    v2f  __attribute__((ext_vector_type(2)));

// ------------------------------ LayerNorm ----------------------------------
__global__ __launch_bounds__(256) void nys_layernorm(
    const float* __restrict__ X, const float* __restrict__ g,
    const float* __restrict__ be, _Float16* __restrict__ o16)
{
  __shared__ float sm[256];
  const int t = threadIdx.x;
  const long long base = (long long)blockIdx.x * 512;
  float x0 = X[base + t], x1 = X[base + 256 + t];
  sm[t] = x0 + x1; __syncthreads();
  for (int s = 128; s > 0; s >>= 1) { if (t < s) sm[t] += sm[t + s]; __syncthreads(); }
  float mu = sm[0] * (1.0f / 512.0f); __syncthreads();
  float d0 = x0 - mu, d1 = x1 - mu;
  sm[t] = d0 * d0 + d1 * d1; __syncthreads();
  for (int s = 128; s > 0; s >>= 1) { if (t < s) sm[t] += sm[t + s]; __syncthreads(); }
  float r = rsqrtf(sm[0] * (1.0f / 512.0f) + 1e-5f);
  o16[base + t]       = (_Float16)(d0 * r * g[t]       + be[t]);
  o16[base + 256 + t] = (_Float16)(d1 * r * g[t + 256] + be[t + 256]);
}

// ------------------------- f16 WMMA GEMM (generic) -------------------------
// Block tile 128x64, 8 waves (4 row x 2 col) each computing 32x32 via 2x2
// v_wmma_f32_16x16x32_f16.  LDS pitch 40 halfs (bank-conflict-free strides),
// double buffered; next tile's global loads are issued before this tile's
// compute so HBM latency hides under the WMMA+LDS work.
// BNMAJOR: 1 if B is stored [N][K] (direct copy), 0 if [K][N] (transpose on
// LDS write).  OUTMODE: 0 standard (D32/D16/bias/Cin/act), 1 fused QKV
// split + q-scale, 2 f16 write to Oc[(b,n),(h*64+dh)] with batch=(b*8+h).
template<int BNMAJOR, int OUTMODE>
__global__ __launch_bounds__(256) void nys_gemm_f16(
    const _Float16* __restrict__ A, long long lda, long long sA,
    const _Float16* __restrict__ B, long long ldb, long long sB,
    const float*    __restrict__ Cin, long long ldc, long long sC,
    float*          __restrict__ D32, long long ldd32, long long sD32,
    _Float16*       __restrict__ D16, long long ldd16, long long sD16,
    const float*    __restrict__ bias,
    int Ktot, int act, float alpha,
    _Float16* __restrict__ Qo, _Float16* __restrict__ Ko, _Float16* __restrict__ Vo)
{
  __shared__ _Float16 As[2][128 * 40];
  __shared__ _Float16 Bs[2][64 * 40];
  const int t = threadIdx.x;
  const int lane = t & 31, wid = t >> 5;
  const int wrow = wid & 3, wcol = wid >> 2;
  const int lanem = lane & 15, laneh = lane >> 4;
  const long long rowBlock = (long long)blockIdx.x * 128;
  const long long colBlock = (long long)blockIdx.y * 64;
  const int batch = blockIdx.z;
  const _Float16* Ab = A + (long long)batch * sA;
  const _Float16* Bb = B + (long long)batch * sB;

  v8f acc[2][2] = {};

  const int ar = t >> 1, aks = (t & 1) * 16;      // A staging: 16 halfs/thread
  const int bn = t >> 2, bks = (t & 3) * 8;       // B staging (N-major)
  const int bk = t >> 3, bns = (t & 7) * 8;       // B staging (K-major, transpose)

  // ---- prologue: stage tile 0 into buffer 0
  {
    const _Float16* asrc = Ab + (rowBlock + ar) * lda + aks;
    v8h a0 = *(const v8h*)asrc;
    v8h a1 = *(const v8h*)(asrc + 8);
    *(v8h*)&As[0][ar * 40 + aks]     = a0;
    *(v8h*)&As[0][ar * 40 + aks + 8] = a1;
    if (BNMAJOR) {
      *(v8h*)&Bs[0][bn * 40 + bks] =
          *(const v8h*)(Bb + (colBlock + bn) * ldb + bks);
    } else {
      v8h bv = *(const v8h*)(Bb + (long long)bk * ldb + colBlock + bns);
#pragma unroll
      for (int i = 0; i < 8; ++i) Bs[0][(bns + i) * 40 + bk] = bv[i];
    }
  }
  __syncthreads();

  int cur = 0;
  for (int k0 = 0; k0 < Ktot; k0 += 32) {
    const bool hasNext = (k0 + 32 < Ktot);
    v8h na0, na1, nb;
    if (hasNext) {
      // ---- issue next tile's global loads before computing (latency hiding)
      const _Float16* asrc = Ab + (rowBlock + ar) * lda + (k0 + 32 + aks);
      na0 = *(const v8h*)asrc;
      na1 = *(const v8h*)(asrc + 8);
      if (BNMAJOR)
        nb = *(const v8h*)(Bb + (colBlock + bn) * ldb + (k0 + 32 + bks));
      else
        nb = *(const v8h*)(Bb + (long long)(k0 + 32 + bk) * ldb + colBlock + bns);
      __builtin_prefetch(asrc + 64, 0, 1);   // speculative: global_prefetch_b8
    }

    // ---- fragments (CDNA5 16-bit A 16x32 / B 32x16 lane layouts) + WMMA
    v16h af[2], bf[2];
    const int kb  = laneh * 8;     // A: k chunks {kb..kb+7} and {kb+16..kb+23}
    const int kb2 = laneh * 8;     // B: k chunks {kb2..kb2+7} and {kb2+16..kb2+23}
#pragma unroll
    for (int mt = 0; mt < 2; ++mt) {
      int r = wrow * 32 + mt * 16 + lanem;
      v8h lo = *(const v8h*)&As[cur][r * 40 + kb];
      v8h hi = *(const v8h*)&As[cur][r * 40 + kb + 16];
#pragma unroll
      for (int i = 0; i < 8; ++i) { af[mt][i] = lo[i]; af[mt][8 + i] = hi[i]; }
    }
#pragma unroll
    for (int nt = 0; nt < 2; ++nt) {
      int c = wcol * 32 + nt * 16 + lanem;
      v8h lo = *(const v8h*)&Bs[cur][c * 40 + kb2];
      v8h hi = *(const v8h*)&Bs[cur][c * 40 + kb2 + 16];
#pragma unroll
      for (int i = 0; i < 8; ++i) { bf[nt][i] = lo[i]; bf[nt][8 + i] = hi[i]; }
    }
#pragma unroll
    for (int mt = 0; mt < 2; ++mt)
#pragma unroll
      for (int nt = 0; nt < 2; ++nt)
        acc[mt][nt] = __builtin_amdgcn_wmma_f32_16x16x32_f16(
            false, af[mt], false, bf[nt], (short)0, acc[mt][nt], false, false);

    if (hasNext) {
      // Safe to write buffer cur^1: the end-of-previous-iteration barrier
      // guarantees all waves finished reading it.
      *(v8h*)&As[cur ^ 1][ar * 40 + aks]     = na0;
      *(v8h*)&As[cur ^ 1][ar * 40 + aks + 8] = na1;
      if (BNMAJOR) {
        *(v8h*)&Bs[cur ^ 1][bn * 40 + bks] = nb;
      } else {
#pragma unroll
        for (int i = 0; i < 8; ++i) Bs[cur ^ 1][(bns + i) * 40 + bk] = nb[i];
      }
      __syncthreads();
    }
    cur ^= 1;
  }

  // ---- epilogue (D layout: VGPR g, lanes0-15 M=g, lanes16-31 M=8+g)
#pragma unroll
  for (int mt = 0; mt < 2; ++mt)
#pragma unroll
    for (int nt = 0; nt < 2; ++nt) {
      long long gc = colBlock + wcol * 32 + nt * 16 + lanem;
      float bv = bias ? bias[gc] : 0.0f;
#pragma unroll
      for (int g = 0; g < 8; ++g) {
        long long gr = rowBlock + wrow * 32 + mt * 16 + laneh * 8 + g;
        float val = acc[mt][nt][g] * alpha + bv;
        if (Cin) val += Cin[(long long)batch * sC + gr * ldc + gc];
        if (act == 1) val = 0.5f * val * (1.0f + erff(val * 0.70710678118654752f));
        if (OUTMODE == 0) {
          if (D32) D32[(long long)batch * sD32 + gr * ldd32 + gc] = val;
          if (D16) D16[(long long)batch * sD16 + gr * ldd16 + gc] = (_Float16)val;
        } else if (OUTMODE == 1) {
          int bb = (int)(gr >> 12), n = (int)(gr & 4095);
          int which = (int)(gc >> 9), h = (int)((gc >> 6) & 7), dh = (int)(gc & 63);
          float s = (which == 0) ? 0.125f : 1.0f;   // q *= DH^-0.5
          _Float16* dst = (which == 0) ? Qo : (which == 1 ? Ko : Vo);
          dst[(((long long)(bb * 8 + h)) * 4096 + n) * 64 + dh] = (_Float16)(val * s);
        } else {  // OUTMODE == 2: (b,h,n,dh) -> Oc[(b*4096+n)*512 + h*64 + dh]
          int bb = batch >> 3, h = batch & 7;
          D16[((long long)bb * 4096 + gr) * 512 + h * 64 + gc] = (_Float16)val;
        }
      }
    }
}

// --------------------- f32 WMMA GEMM, batched 256x256x256 ------------------
// D = alpha * A @ B per batch (A,B,D all 256x256 row-major, stride 65536).
// Same double-buffered pipeline, v_wmma_f32_16x16x4_f32.
__global__ __launch_bounds__(256) void nys_gemm256_f32(
    const float* __restrict__ A, const float* __restrict__ B,
    float* __restrict__ D, float alpha)
{
  __shared__ float Asf[2][128 * 36];
  __shared__ float Bsf[2][64 * 36];
  const int t = threadIdx.x;
  const int lane = t & 31, wid = t >> 5;
  const int wrow = wid & 3, wcol = wid >> 2;
  const int lanem = lane & 15, laneh = lane >> 4;
  const long long rowBlock = (long long)blockIdx.x * 128;
  const long long colBlock = (long long)blockIdx.y * 64;
  const long long base = (long long)blockIdx.z * 65536;

  v8f acc[2][2] = {};
  const int ar = t >> 1, aks = (t & 1) * 16;
  const int bk = t >> 3, bns = (t & 7) * 8;

  {
    const float* asrc = A + base + (rowBlock + ar) * 256 + aks;
#pragma unroll
    for (int i = 0; i < 4; ++i)
      *(float4*)&Asf[0][ar * 36 + aks + 4 * i] = ((const float4*)asrc)[i];
    const float* bsrc = B + base + (long long)bk * 256 + colBlock + bns;
    float4 b0 = ((const float4*)bsrc)[0];
    float4 b1 = ((const float4*)bsrc)[1];
    float bv[8] = { b0.x, b0.y, b0.z, b0.w, b1.x, b1.y, b1.z, b1.w };
#pragma unroll
    for (int i = 0; i < 8; ++i) Bsf[0][(bns + i) * 36 + bk] = bv[i];
  }
  __syncthreads();

  int cur = 0;
  for (int k0 = 0; k0 < 256; k0 += 32) {
    const bool hasNext = (k0 + 32 < 256);
    float4 na[4], nb0, nb1;
    if (hasNext) {
      const float* asrc = A + base + (rowBlock + ar) * 256 + k0 + 32 + aks;
#pragma unroll
      for (int i = 0; i < 4; ++i) na[i] = ((const float4*)asrc)[i];
      const float* bsrc = B + base + (long long)(k0 + 32 + bk) * 256 + colBlock + bns;
      nb0 = ((const float4*)bsrc)[0];
      nb1 = ((const float4*)bsrc)[1];
      __builtin_prefetch(asrc + 64, 0, 1);
    }

#pragma unroll
    for (int kk = 0; kk < 8; ++kk) {         // 8 x (16x16x4) per 32-k stage
      v2f av[2], bvf[2];
#pragma unroll
      for (int mt = 0; mt < 2; ++mt) {
        int r = wrow * 32 + mt * 16 + lanem;
        float2 x = *(const float2*)&Asf[cur][r * 36 + kk * 4 + laneh * 2];
        av[mt][0] = x.x; av[mt][1] = x.y;
      }
#pragma unroll
      for (int nt = 0; nt < 2; ++nt) {
        int c = wcol * 32 + nt * 16 + lanem;
        float2 x = *(const float2*)&Bsf[cur][c * 36 + kk * 4 + laneh * 2];
        bvf[nt][0] = x.x; bvf[nt][1] = x.y;
      }
#pragma unroll
      for (int mt = 0; mt < 2; ++mt)
#pragma unroll
        for (int nt = 0; nt < 2; ++nt)
          acc[mt][nt] = __builtin_amdgcn_wmma_f32_16x16x4_f32(
              false, av[mt], false, bvf[nt], (short)0, acc[mt][nt], false, false);
    }

    if (hasNext) {
#pragma unroll
      for (int i = 0; i < 4; ++i)
        *(float4*)&Asf[cur ^ 1][ar * 36 + aks + 4 * i] = na[i];
      float bv[8] = { nb0.x, nb0.y, nb0.z, nb0.w, nb1.x, nb1.y, nb1.z, nb1.w };
#pragma unroll
      for (int i = 0; i < 8; ++i) Bsf[cur ^ 1][(bns + i) * 36 + bk] = bv[i];
      __syncthreads();
    }
    cur ^= 1;
  }

#pragma unroll
  for (int mt = 0; mt < 2; ++mt)
#pragma unroll
    for (int nt = 0; nt < 2; ++nt) {
      long long gc = colBlock + wcol * 32 + nt * 16 + lanem;
#pragma unroll
      for (int g = 0; g < 8; ++g) {
        long long gr = rowBlock + wrow * 32 + mt * 16 + laneh * 8 + g;
        D[base + gr * 256 + gc] = alpha * acc[mt][nt][g];
      }
    }
}

// ------------------------------ softmax rows -------------------------------
__global__ __launch_bounds__(256) void nys_softmax_rows(
    const float* __restrict__ S, _Float16* __restrict__ o16,
    float* __restrict__ o32, int L)
{
  __shared__ float sm[256];
  const int t = threadIdx.x;
  const long long base = (long long)blockIdx.x * L;
  float m = -1e30f;
  for (int i = t; i < L; i += 256) m = fmaxf(m, S[base + i]);
  sm[t] = m; __syncthreads();
  for (int s = 128; s > 0; s >>= 1) { if (t < s) sm[t] = fmaxf(sm[t], sm[t + s]); __syncthreads(); }
  m = sm[0]; __syncthreads();
  float sum = 0.0f;
  for (int i = t; i < L; i += 256) sum += expf(S[base + i] - m);
  sm[t] = sum; __syncthreads();
  for (int s = 128; s > 0; s >>= 1) { if (t < s) sm[t] += sm[t + s]; __syncthreads(); }
  float inv = 1.0f / sm[0];
  for (int i = t; i < L; i += 256) {
    float e = expf(S[base + i] - m) * inv;
    if (o16) o16[base + i] = (_Float16)e;
    if (o32) o32[base + i] = e;          // in-place safe (per-thread indices)
  }
}

// --------------------------- landmark means (l=16) -------------------------
__global__ void nys_landmarks(const _Float16* __restrict__ q,
                              const _Float16* __restrict__ k,
                              _Float16* __restrict__ ql, _Float16* __restrict__ kl)
{
  long long i = (long long)blockIdx.x * 256 + threadIdx.x;   // 1,048,576
  int dh = (int)(i & 63); int m = (int)((i >> 6) & 255); long long bh = i >> 14;
  long long rowbase = (bh * 4096 + (long long)m * 16) * 64 + dh;
  float sq = 0.f, sk = 0.f;
#pragma unroll
  for (int j = 0; j < 16; ++j) {
    sq += (float)q[rowbase + j * 64];
    sk += (float)k[rowbase + j * 64];
  }
  long long o = (bh * 256 + m) * 64 + dh;
  ql[o] = (_Float16)(sq * (1.0f / 16.0f));
  kl[o] = (_Float16)(sk * (1.0f / 16.0f));
}

// -------------------- depthwise conv residual over sequence ----------------
__global__ void nys_conv_res(const _Float16* __restrict__ v,
                             const float* __restrict__ w, float* __restrict__ O)
{
  long long i = (long long)blockIdx.x * 256 + threadIdx.x;   // 16,777,216
  int dh = (int)(i & 63); int n = (int)((i >> 6) & 4095);
  long long bh = i >> 18; int h = (int)(bh & 7);
  const float* wh = w + h * 33;
  long long base = bh * 4096 * 64 + dh;
  float acc = 0.0f;
#pragma unroll
  for (int j = 0; j < 33; ++j) {
    int nn = n + j - 16;
    if (nn >= 0 && nn < 4096) acc += wh[j] * (float)v[base + (long long)nn * 64];
  }
  O[i] = acc;
}

// ---------------- pinv helpers: global abs row/col sum max -----------------
__global__ void nys_init_scal(unsigned* s) { if (threadIdx.x < 2) s[threadIdx.x] = 0u; }

__global__ __launch_bounds__(256) void nys_abs_rowcol_max(
    const float* __restrict__ x, unsigned* __restrict__ scal)
{
  __shared__ float sm[256];
  const int t = threadIdx.x;
  long long bh = blockIdx.x >> 8; int r = blockIdx.x & 255;
  long long base = bh << 16;
  float v = (blockIdx.y == 0) ? fabsf(x[base + (long long)r * 256 + t])
                              : fabsf(x[base + (long long)t * 256 + r]);
  sm[t] = v; __syncthreads();
  for (int s = 128; s > 0; s >>= 1) { if (t < s) sm[t] += sm[t + s]; __syncthreads(); }
  if (t == 0) atomicMax(&scal[blockIdx.y], __float_as_uint(sm[0]));  // sums >= 0
}

__global__ void nys_zinit(const float* __restrict__ x,
                          const unsigned* __restrict__ scal, float* __restrict__ z)
{
  long long i = (long long)blockIdx.x * 256 + threadIdx.x;   // 4,194,304
  float inv = 1.0f / (__uint_as_float(scal[0]) * __uint_as_float(scal[1]));
  long long bh = i >> 16; int r = (int)((i >> 8) & 255); int c = (int)(i & 255);
  z[i] = x[(bh << 16) + (long long)c * 256 + r] * inv;       // x^T * inv
}

__global__ void nys_diag_sub(const float* __restrict__ s, float* __restrict__ o, float c)
{
  long long i = (long long)blockIdx.x * 256 + threadIdx.x;
  int r = (int)((i >> 8) & 255), cc = (int)(i & 255);
  o[i] = (r == cc ? c : 0.0f) - s[i];
}

__global__ void nys_f32_to_f16(const float* __restrict__ s, _Float16* __restrict__ d)
{
  long long i = (long long)blockIdx.x * 256 + threadIdx.x;
  d[i] = (_Float16)s[i];
}

// ---------------------------------------------------------------------------
extern "C" void kernel_launch(void* const* d_in, const int* in_sizes, int n_in,
                              void* d_out, int out_size, void* d_ws, size_t ws_size,
                              hipStream_t stream)
{
  const float* X     = (const float*)d_in[0];
  const float* Wqkv  = (const float*)d_in[1];
  const float* Wout  = (const float*)d_in[2];
  const float* bout  = (const float*)d_in[3];
  const float* convw = (const float*)d_in[4];
  const float* g1    = (const float*)d_in[5];
  const float* be1   = (const float*)d_in[6];
  const float* g2    = (const float*)d_in[7];
  const float* be2   = (const float*)d_in[8];
  const float* W1    = (const float*)d_in[9];
  const float* b1    = (const float*)d_in[10];
  const float* W2    = (const float*)d_in[11];
  const float* b2    = (const float*)d_in[12];
  float* OUT = (float*)d_out;
  (void)in_sizes; (void)n_in; (void)out_size; (void)ws_size;

  char* ws = (char*)d_ws;
  size_t cur = 0;
  auto alloc = [&](size_t bytes) -> char* {
    char* p = ws + cur; cur = (cur + bytes + 255) & ~(size_t)255; return p;
  };

  // Lifetime-aliased workspace (peak ~790 MB).
  _Float16* hln   = (_Float16*)alloc(16777216ull * 2);  // hln -> Oc -> ln2in
  _Float16* q16   = (_Float16*)alloc(16777216ull * 2);
  _Float16* k16   = (_Float16*)alloc(16777216ull * 2);
  _Float16* v16   = (_Float16*)alloc(16777216ull * 2);
  _Float16* ql16  = (_Float16*)alloc(131072ull * 2);
  _Float16* kl16  = (_Float16*)alloc(131072ull * 2);
  float*    bigA  = (float*)   alloc(67108864ull * 4);  // sim3 f32 -> sim1 f32 -> out1 f16
  _Float16* bigB  = (_Float16*)alloc(67108864ull * 2);  // attn3 -> attn1 -> gelu acts
  float*    S2f   = (float*)   alloc(4194304ull * 4);   // sim2 / attn2 (in place)
  float*    Zp    = (float*)   alloc(4194304ull * 4);
  float*    Zq    = (float*)   alloc(4194304ull * 4);
  float*    XZf   = (float*)   alloc(4194304ull * 4);
  float*    Uf    = (float*)   alloc(4194304ull * 4);
  float*    Tf    = (float*)   alloc(4194304ull * 4);
  _Float16* z16   = (_Float16*)alloc(4194304ull * 2);
  _Float16* kv16  = (_Float16*)alloc(1048576ull * 2);
  float*    Obuf  = (float*)   alloc(16777216ull * 4);  // conv residual (b,h,n,dh)
  float*    htr   = (float*)   alloc(16777216ull * 4);  // attention output trunk
  _Float16* wqkv16= (_Float16*)alloc(786432ull * 2);
  _Float16* wout16= (_Float16*)alloc(262144ull * 2);
  _Float16* w116  = (_Float16*)alloc(1048576ull * 2);
  _Float16* w216  = (_Float16*)alloc(1048576ull * 2);
  unsigned* scal  = (unsigned*)alloc(256);
  _Float16* Oc    = hln;                                // alias
  _Float16* out1  = (_Float16*)bigA;                    // alias (f16 view)
  _Float16* gbuf  = bigB;                               // alias

  // 0) weight conversions to f16
  nys_f32_to_f16<<<786432 / 256, 256, 0, stream>>>(Wqkv, wqkv16);
  nys_f32_to_f16<<<262144 / 256, 256, 0, stream>>>(Wout, wout16);
  nys_f32_to_f16<<<1048576 / 256, 256, 0, stream>>>(W1, w116);
  nys_f32_to_f16<<<1048576 / 256, 256, 0, stream>>>(W2, w216);

  // 1) LN1
  nys_layernorm<<<32768, 256, 0, stream>>>(X, g1, be1, hln);

  // 2) QKV GEMM (32768x1536, K=512), fused q-scale + split into (b,h,n,dh)
  nys_gemm_f16<0, 1><<<dim3(256, 24, 1), 256, 0, stream>>>(
      hln, 512, 0, wqkv16, 1536, 0,
      nullptr, 0, 0, nullptr, 0, 0, nullptr, 0, 0, nullptr,
      512, 0, 1.0f, q16, k16, v16);

  // 3) landmark means
  nys_landmarks<<<4096, 256, 0, stream>>>(q16, k16, ql16, kl16);

  // 4) sim3 = q_l @ k^T  (per bh: 256x4096, K=64) -> softmax -> attn3 (f16)
  nys_gemm_f16<1, 0><<<dim3(2, 64, 64), 256, 0, stream>>>(
      ql16, 64, 16384, k16, 64, 262144,
      nullptr, 0, 0, bigA, 4096, 1048576, nullptr, 0, 0, nullptr,
      64, 0, 1.0f, nullptr, nullptr, nullptr);
  nys_softmax_rows<<<16384, 256, 0, stream>>>(bigA, bigB, nullptr, 4096);

  // 5) kv = attn3 @ v  (per bh: 256x64, K=4096) -> f16
  nys_gemm_f16<0, 0><<<dim3(2, 1, 64), 256, 0, stream>>>(
      bigB, 4096, 1048576, v16, 64, 262144,
      nullptr, 0, 0, nullptr, 0, 0, kv16, 64, 16384, nullptr,
      4096, 0, 1.0f, nullptr, nullptr, nullptr);

  // 6) sim2 = q_l @ k_l^T (256x256) -> softmax (f32, in place)
  nys_gemm_f16<1, 0><<<dim3(2, 4, 64), 256, 0, stream>>>(
      ql16, 64, 16384, kl16, 64, 16384,
      nullptr, 0, 0, S2f, 256, 65536, nullptr, 0, 0, nullptr,
      64, 0, 1.0f, nullptr, nullptr, nullptr);
  nys_softmax_rows<<<16384, 256, 0, stream>>>(S2f, nullptr, S2f, 256);

  // 7) Moore-Penrose pinv (f32 WMMA), z0 = x^T / (max(colsum)*max(rowsum))
  nys_init_scal<<<1, 32, 0, stream>>>(scal);
  nys_abs_rowcol_max<<<dim3(16384, 2, 1), 256, 0, stream>>>(S2f, scal);
  nys_zinit<<<16384, 256, 0, stream>>>(S2f, scal, Zp);
  {
    dim3 g256(2, 4, 64);
    float* zc = Zp; float* zn = Zq;
    for (int it = 0; it < 6; ++it) {
      nys_gemm256_f32<<<g256, 256, 0, stream>>>(S2f, zc, XZf, 1.0f);   // xz = x@z
      nys_diag_sub<<<16384, 256, 0, stream>>>(XZf, Tf, 7.0f);          // 7I - xz
      nys_gemm256_f32<<<g256, 256, 0, stream>>>(XZf, Tf, Uf, 1.0f);
      nys_diag_sub<<<16384, 256, 0, stream>>>(Uf, Tf, 15.0f);          // 15I - .
      nys_gemm256_f32<<<g256, 256, 0, stream>>>(XZf, Tf, Uf, 1.0f);
      nys_diag_sub<<<16384, 256, 0, stream>>>(Uf, Tf, 13.0f);          // 13I - .
      nys_gemm256_f32<<<g256, 256, 0, stream>>>(zc, Tf, zn, 0.25f);    // z = .25 z@.
      float* tmp = zc; zc = zn; zn = tmp;
    }
    nys_f32_to_f16<<<16384, 256, 0, stream>>>(zc, z16);
  }

  // 8) sim1 = q @ k_l^T (per bh: 4096x256, K=64) -> softmax -> attn1 (f16)
  nys_gemm_f16<1, 0><<<dim3(32, 4, 64), 256, 0, stream>>>(
      q16, 64, 262144, kl16, 64, 16384,
      nullptr, 0, 0, bigA, 256, 1048576, nullptr, 0, 0, nullptr,
      64, 0, 1.0f, nullptr, nullptr, nullptr);
  nys_softmax_rows<<<262144, 256, 0, stream>>>(bigA, bigB, nullptr, 256);

  // 9) out1 = attn1 @ attn2_inv (4096x256, K=256) -> f16 (into bigA region)
  nys_gemm_f16<0, 0><<<dim3(32, 4, 64), 256, 0, stream>>>(
      bigB, 256, 1048576, z16, 256, 65536,
      nullptr, 0, 0, nullptr, 0, 0, out1, 256, 1048576, nullptr,
      256, 0, 1.0f, nullptr, nullptr, nullptr);

  // 10) depthwise conv residual -> Obuf (b,h,n,dh)
  nys_conv_res<<<65536, 256, 0, stream>>>(v16, convw, Obuf);

  // 11) out = out1 @ kv + res, written transposed to Oc[(b,n),(h,dh)]
  nys_gemm_f16<0, 2><<<dim3(32, 1, 64), 256, 0, stream>>>(
      out1, 256, 1048576, kv16, 64, 16384,
      Obuf, 64, 262144, nullptr, 0, 0, Oc, 0, 0, nullptr,
      256, 0, 1.0f, nullptr, nullptr, nullptr);

  // 12) attention output projection: htr = Oc @ Wout + bout  (f32 trunk)
  nys_gemm_f16<0, 0><<<dim3(256, 8, 1), 256, 0, stream>>>(
      Oc, 512, 0, wout16, 512, 0,
      nullptr, 0, 0, htr, 512, 0, nullptr, 0, 0, bout,
      512, 0, 1.0f, nullptr, nullptr, nullptr);

  // 13) LN2 (reuses hln/Oc buffer for the f16 normalized activations)
  nys_layernorm<<<32768, 256, 0, stream>>>(htr, g2, be2, hln);

  // 14) MLP1: gelu(ln2 @ W1 + b1) -> f16 (bigB region)
  nys_gemm_f16<0, 0><<<dim3(256, 32, 1), 256, 0, stream>>>(
      hln, 512, 0, w116, 2048, 0,
      nullptr, 0, 0, nullptr, 0, 0, gbuf, 2048, 0, b1,
      512, 1, 1.0f, nullptr, nullptr, nullptr);

  // 15) MLP2 + residual: OUT = htr + (gbuf @ W2 + b2)
  nys_gemm_f16<0, 0><<<dim3(256, 8, 1), 256, 0, stream>>>(
      gbuf, 2048, 0, w216, 512, 0,
      htr, 512, 0, OUT, 512, 0, nullptr, 0, 0, b2,
      2048, 0, 1.0f, nullptr, nullptr, nullptr);
}
